// RAMAttention_89489938579811
// MI455X (gfx1250) — 8-bit matrix operand order repaired and checked by reading, compiled, not run on hardware
//
#include <hip/hip_runtime.h>
#include <hip/hip_bf16.h>

typedef __attribute__((ext_vector_type(8))) int v8i;

#define Bn  256
#define Sn  64
#define IBn 32
#define Hn  4

__global__ __launch_bounds__(256)
void ram_attention_kernel(const int* __restrict__ tokens,
                          const int* __restrict__ sim_conn,
                          const float* __restrict__ sim_mem,
                          const int* __restrict__ val_conn,
                          const float* __restrict__ val_mem,
                          const float* __restrict__ agg_mem,
                          const int* __restrict__ out_conn,
                          const float* __restrict__ out_mem,
                          float* __restrict__ out)
{
    __shared__ unsigned tok_bits[Sn];
    __shared__ unsigned ak_l[Hn * Sn];
    __shared__ unsigned sim_bits[(Hn * 4096) / 32];
    __shared__ unsigned val_bits[(Hn * IBn * 1024) / 32];
    __shared__ unsigned Pbits[Hn * Sn];
    __shared__ unsigned att_lo[Hn * Sn];
    __shared__ unsigned att_hi[Hn * Sn];
    __shared__ unsigned char nz_l[Hn * Sn];
    __shared__ unsigned char agg_l[Hn * IBn * (Sn + 1)];
    __shared__ unsigned char comb[Sn * Hn * IBn];
    __shared__ int sconn_l[Hn * 12];
    __shared__ int vconn_l[Hn * IBn * 10];
    __shared__ int oconn_l[IBn * 12];

    const int tid  = threadIdx.x;
    const int b    = blockIdx.x;
    const int wv   = tid >> 5;
    const int lane = tid & 31;

    if (tid < Sn) {
        unsigned tb = 0;
        const int* tr = tokens + ((size_t)b * Sn + tid) * IBn;
        #pragma unroll
        for (int k = 0; k < IBn; ++k) tb |= ((unsigned)(tr[k] & 1)) << k;
        tok_bits[tid] = tb;
    }
    if (tid < Hn * 12) sconn_l[tid] = sim_conn[tid];
    for (int f = tid; f < Hn * IBn * 10; f += 256) vconn_l[f] = val_conn[f];
    for (int f = tid; f < IBn * 12; f += 256) oconn_l[f] = out_conn[f];

    for (int w = wv; w < (Hn * 4096) / 32; w += 8) {
        unsigned bal = __builtin_amdgcn_ballot_w32(sim_mem[w * 32 + lane] > 0.5f);
        if (lane == 0) sim_bits[w] = bal;
    }
    for (int w = wv; w < (Hn * IBn * 1024) / 32; w += 8) {
        unsigned bal = __builtin_amdgcn_ballot_w32(val_mem[w * 32 + lane] > 0.5f);
        if (lane == 0) val_bits[w] = bal;
    }
    for (int f = tid; f < Hn * IBn * (Sn + 1); f += 256)
        agg_l[f] = (unsigned char)(agg_mem[f] > 0.5f);
    __syncthreads();

    const int h = tid >> 6;
    const int i = tid & 63;
    unsigned aq = 0, akv = 0;
    #pragma unroll
    for (int t = 0; t < 12; ++t) {
        int k = sconn_l[h * 12 + t];
        unsigned qb = 0, kb = 0;
        if (k < 32)       qb = (tok_bits[i] >> k) & 1u;
        else if (k < 64)  kb = (tok_bits[i] >> (k - 32)) & 1u;
        else if (k < 70)  qb = ((unsigned)i >> (69 - k)) & 1u;
        else              kb = ((unsigned)i >> (75 - k)) & 1u;
        aq  = (aq  << 1) | qb;
        akv = (akv << 1) | kb;
    }
    ak_l[h * 64 + i] = akv;
    __syncthreads();

    {
        unsigned lo = 0, hiw = 0;
        for (int j = 0; j <= i; ++j) {
            unsigned addr = aq | ak_l[h * 64 + j];
            unsigned bit = (sim_bits[(h << 7) + (addr >> 5)] >> (addr & 31)) & 1u;
            if (j < 32) lo |= bit << j; else hiw |= bit << (j - 32);
        }
        att_lo[h * 64 + i] = lo;
        att_hi[h * 64 + i] = hiw;
        nz_l[h * 64 + i] = (lo | hiw) ? (unsigned char)0xFF : (unsigned char)0;

        unsigned pbw = 0;
        for (int n = 0; n < IBn; ++n) {
            int addr = 0;
            const int* vc = &vconn_l[(h * IBn + n) * 10];
            #pragma unroll
            for (int t = 0; t < 10; ++t) {
                int k = vc[t];
                unsigned bit = (k < 32) ? ((tok_bits[i] >> k) & 1u)
                                        : (((unsigned)i >> (37 - k)) & 1u);
                addr = (addr << 1) | (int)bit;
            }
            unsigned f = (unsigned)((h * IBn + n) * 1024 + addr);
            pbw |= ((val_bits[f >> 5] >> (f & 31)) & 1u) << n;
        }
        Pbits[h * 64 + i] = pbw;
    }
    __syncthreads();

    {
        const int halfw = lane >> 4;
        const int lm    = lane & 15;
        const int hh    = wv >> 1;
        const int nt    = wv & 1;
        const int no    = 16 * nt + lm;
        const int aggb  = (hh * IBn + no) * (Sn + 1);

        v8i bmat;
        #pragma unroll
        for (int v = 0; v < 8; ++v) {
            int kbase = 32 * (v >> 2) + 16 * halfw + 4 * (v & 3);
            unsigned d = 0;
            #pragma unroll
            for (int q = 0; q < 4; ++q)
                d |= ((Pbits[hh * 64 + kbase + q] >> no) & 1u) << (8 * q);
            bmat[v] = (int)d;
        }

        #pragma unroll
        for (int mt = 0; mt < 4; ++mt) {
            int ia = 16 * mt + lm;
            unsigned rlo = att_lo[hh * 64 + ia];
            unsigned rhi = att_hi[hh * 64 + ia];
            v8i amat;
            #pragma unroll
            for (int v = 0; v < 8; ++v) {
                int jb = 32 * (v >> 2) + 16 * halfw + 4 * (v & 3);
                unsigned word = (jb >= 32) ? rhi : rlo;
                unsigned nib = (word >> (jb & 31)) & 0xFu;
                amat[v] = (int)((nib * 0x00204081u) & 0x01010101u);
            }
            v8i cz = {};
            v8i dacc = __builtin_amdgcn_wmma_i32_16x16x64_iu8(
                false, amat, false, bmat, cz, false, false);

            #pragma unroll
            for (int r = 0; r < 8; ++r) {
                int c = dacc[r];
                c = (c < 0) ? 0 : ((c > Sn) ? Sn : c);
                int io = 16 * mt + r + 8 * halfw;
                unsigned char a = agg_l[aggb + c];
                comb[io * (Hn * IBn) + hh * IBn + no] = (unsigned char)(a & nz_l[hh * 64 + io]);
            }
        }
    }
    __syncthreads();

    for (int q = 0; q < 8; ++q) {
        int o = tid + 256 * q;
        int s = o >> 5;
        int n = o & 31;
        int addr = 0;
        const int* oc = &oconn_l[n * 12];
        #pragma unroll
        for (int t = 0; t < 12; ++t)
            addr = (addr << 1) | (int)(comb[s * (Hn * IBn) + oc[t]] & 1);
        out[((size_t)b * Sn + s) * IBn + n] = out_mem[n * 4096 + addr];
    }
}

extern "C" void kernel_launch(void* const* d_in, const int* in_sizes, int n_in,
                              void* d_out, int out_size, void* d_ws, size_t ws_size,
                              hipStream_t stream) {
    (void)in_sizes; (void)n_in; (void)out_size; (void)d_ws; (void)ws_size;
    const int*   tokens   = (const int*)  d_in[0];
    const int*   sim_conn = (const int*)  d_in[1];
    const float* sim_mem  = (const float*)d_in[2];
    const int*   val_conn = (const int*)  d_in[3];
    const float* val_mem  = (const float*)d_in[4];
    const float* agg_mem  = (const float*)d_in[5];
    const int*   out_conn = (const int*)  d_in[6];
    const float* out_mem  = (const float*)d_in[7];

    ram_attention_kernel<<<dim3(Bn), dim3(256), 0, stream>>>(
        tokens, sim_conn, sim_mem, val_conn, val_mem, agg_mem, out_conn, out_mem,
        (float*)d_out);
}
